// GaitEventModel_83485574300356
// MI455X (gfx1250) — matrix operand order repaired and checked by reading, not compiled, not run on hardware
//
#include <hip/hip_runtime.h>

// ---------------- types for WMMA ----------------
typedef __bf16 bf16_t;
typedef bf16_t v16bf __attribute__((ext_vector_type(16)));
typedef bf16_t v8bf  __attribute__((ext_vector_type(8)));   // 16 bytes
typedef float  v8f   __attribute__((ext_vector_type(8)));

#define BSZ   64
#define TLEN  2048
#define HDIM  128
#define G3    384          // 3*H
#define BT    (BSZ * TLEN)

#define WMMA_BF16(a, b, c) \
    __builtin_amdgcn_wmma_f32_16x16x32_bf16(false, (a), false, (b), (short)0, (c), false, false)

// ---- A fragment: 16x32 bf16, rows of matrix A (M dim) ----
// lane L: M = L&15 ; element e: K = (L>=16)*8 + (e>=8)*16 + (e&7)
__device__ __forceinline__ v16bf load_frag_a(const bf16_t* base, int row, int ld, int kbase) {
    int lane = threadIdx.x & 31;
    int m    = lane & 15;
    int ksel = (lane >> 4) << 3;                 // 0 or 8
    const bf16_t* p = base + (size_t)(row + m) * ld + kbase + ksel;
    union { v16bf v; v8bf h[2]; } u;
    u.h[0] = *(const v8bf*)(p);                  // K = kbase+ksel .. +7
    u.h[1] = *(const v8bf*)(p + 16);             // K = kbase+ksel+16 .. +23
    return u.v;
}

// ---- B fragment: 32x16 bf16 where B[k][n] = W[n][k] (row-major W, stride ld) ----
// lane L: N = L&15 ; element e: K = kbase + (L>=16)*8 + (e>=8)*16 + (e&7)
__device__ __forceinline__ v16bf load_frag_b(const bf16_t* base, int col, int ld, int kbase) {
    int lane = threadIdx.x & 31;
    int n    = lane & 15;
    int koff = (lane >> 4) << 3;                 // 0 or 8
    const bf16_t* p = base + (size_t)(col + n) * ld + kbase + koff;
    union { v16bf v; v8bf h[2]; } u;
    u.h[0] = *(const v8bf*)(p);
    u.h[1] = *(const v8bf*)(p + 16);
    return u.v;
}

__device__ __forceinline__ float fast_sigmoid(float x) {
    return 1.f / (1.f + __expf(-x));
}
__device__ __forceinline__ float fast_tanh(float x) {
    float e = __expf(2.f * x);
    return 1.f - 2.f / (e + 1.f);
}

// ---------------- weight conversion fp32 -> bf16 ----------------
__global__ void gru_convert_kernel(const float* __restrict__ whh_l0,  const float* __restrict__ whh_l0r,
                                   const float* __restrict__ whh_l1,  const float* __restrict__ whh_l1r,
                                   const float* __restrict__ wih_l1,  const float* __restrict__ wih_l1r,
                                   bf16_t* __restrict__ whh0_bf,      // [2][384][128]
                                   bf16_t* __restrict__ whh1_bf,      // [2][384][128]
                                   bf16_t* __restrict__ wih1_bf)      // [2][384][256]
{
    int i = blockIdx.x * blockDim.x + threadIdx.x;
    int stride = gridDim.x * blockDim.x;
    for (int k = i; k < G3 * HDIM; k += stride) {
        whh0_bf[k]             = (bf16_t)whh_l0[k];
        whh0_bf[G3 * HDIM + k] = (bf16_t)whh_l0r[k];
        whh1_bf[k]             = (bf16_t)whh_l1[k];
        whh1_bf[G3 * HDIM + k] = (bf16_t)whh_l1r[k];
    }
    for (int k = i; k < G3 * 256; k += stride) {
        wih1_bf[k]            = (bf16_t)wih_l1[k];
        wih1_bf[G3 * 256 + k] = (bf16_t)wih_l1r[k];
    }
}

// ---------------- layer-0 input projection (K=18, VALU) ----------------
// xg layout: [dir][t][384][64] (batch innermost, for vector loads in the scan)
// xg = x@Wih^T + b_ih + (g<256 ? b_hh : 0)   (b_hh_n stays in the recurrence)
__global__ __launch_bounds__(128)
void gru_proj0_kernel(const float* __restrict__ x,      // [BT][18], bt = b*T + t
                      const float* __restrict__ w_f,    // [384][18]
                      const float* __restrict__ w_r,
                      const float* __restrict__ bi_f,
                      const float* __restrict__ bi_r,
                      const float* __restrict__ bh_f,
                      const float* __restrict__ bh_r,
                      bf16_t* __restrict__ xg)          // [2][T][384][64]
{
    __shared__ float sx[18];
    int bt  = blockIdx.x;
    int b   = bt >> 11;        // T = 2048
    int t   = bt & (TLEN - 1);
    int tid = threadIdx.x;
    if (tid < 18) sx[tid] = x[(size_t)bt * 18 + tid];
    __syncthreads();
    for (int dir = 0; dir < 2; ++dir) {
        const float* w  = dir ? w_r  : w_f;
        const float* bi = dir ? bi_r : bi_f;
        const float* bh = dir ? bh_r : bh_f;
        bf16_t* out = xg + ((size_t)dir * TLEN + t) * (G3 * BSZ) + b;
        for (int g = tid; g < G3; g += 128) {
            float acc = bi[g] + (g < 256 ? bh[g] : 0.f);
            const float* wr = w + g * 18;
#pragma unroll
            for (int d = 0; d < 18; ++d) acc += sx[d] * wr[d];
            out[(size_t)g * BSZ] = (bf16_t)acc;
        }
    }
}

// ---------------- layer-1 input projection: WMMA GEMM ----------------
__global__ __launch_bounds__(256)
void gru_proj1_kernel(const bf16_t* __restrict__ h0,   // [BT][256]
                      const bf16_t* __restrict__ wih,  // [2][384][256]
                      const float* __restrict__ bi_f,
                      const float* __restrict__ bi_r,
                      const float* __restrict__ bh_f,
                      const float* __restrict__ bh_r,
                      bf16_t* __restrict__ xg)         // [2][T][384][64]
{
    int row0 = blockIdx.x * 64;
    int dir  = blockIdx.y / 6;
    int g0   = (blockIdx.y % 6) * 64;
    const bf16_t* w  = wih + (size_t)dir * G3 * 256;
    const float*  bi = dir ? bi_r : bi_f;
    const float*  bh = dir ? bh_r : bh_f;
    bf16_t* out = xg + (size_t)dir * TLEN * G3 * BSZ;
    int wave = threadIdx.x >> 5;
    int lane = threadIdx.x & 31;
#pragma unroll
    for (int i = 0; i < 2; ++i) {
        int tile = wave + i * 8;            // 16 tiles (4 mt x 4 nt)
        int mt = tile >> 2, nt = tile & 3;
        v8f c = {0.f, 0.f, 0.f, 0.f, 0.f, 0.f, 0.f, 0.f};
#pragma unroll
        for (int kk = 0; kk < 8; ++kk) {
            v16bf a = load_frag_a(h0, row0 + mt * 16, 256, kk * 32);
            v16bf b = load_frag_b(w,  g0   + nt * 16, 256, kk * 32);
            c = WMMA_BF16(a, b, c);
        }
        int n  = lane & 15;
        int mh = (lane >> 4) << 3;
        int g  = g0 + nt * 16 + n;
        float bv = bi[g] + (g < 256 ? bh[g] : 0.f);
#pragma unroll
        for (int v = 0; v < 8; ++v) {
            int row = row0 + mt * 16 + mh + v;   // bt = b*T + t
            int b_  = row >> 11;
            int t_  = row & (TLEN - 1);
            out[((size_t)t_ * G3 + g) * BSZ + b_] = (bf16_t)(c[v] + bv);
        }
    }
}

// ---------------- GRU recurrence: one workgroup per direction ----------------
// 512 threads = 16 waves. wave w: hidden-column slice c = w>>1 (16 cols),
// M-half mh2 = w&1 (rows 32*mh2 .. 32*mh2+31 as two 16-row tiles).
// w_hh fragments (r/z/n x 4 K-steps) live in VGPRs for the whole scan.
// h_prev lives in VGPRs; only the bf16 shadow of h (WMMA A operand) is in LDS.
// xg is [t][384][64] so each gate chunk's 8 batch rows = one aligned 16B load.
__global__ __launch_bounds__(512)
void gru_rec_kernel(const bf16_t* __restrict__ xg,    // [2][T][384][64]
                    const bf16_t* __restrict__ whh,   // [2][384][128] bf16
                    const float* __restrict__ bhh_f,  // [384]
                    const float* __restrict__ bhh_r,  // [384]
                    bf16_t* __restrict__ hout)        // [BT][256], feature offset dir*128
{
    __shared__ bf16_t sHbf[BSZ * HDIM];               // 16 KB

    const int dir = blockIdx.x;
    const int tid = threadIdx.x;
    const int wave = tid >> 5;
    const int lane = tid & 31;
    const int c    = wave >> 1;                        // 0..7 : hidden column group
    const int mh2  = wave & 1;                         // 0..1 : M half
    const int n    = lane & 15;
    const int mh   = (lane >> 4) << 3;                 // 0 or 8
    const int j    = c * 16 + n;                       // this lane's hidden column

    const bf16_t* xgd  = xg  + (size_t)dir * TLEN * G3 * BSZ;
    const bf16_t* whhd = whh + (size_t)dir * G3 * HDIM;
    const float   bhhn = (dir ? bhh_r : bhh_f)[256 + j];   // only n-gate bias remains

    // ---- persistent B fragments: w_hh^T tiles for gates r (nt=c), z (8+c), n (16+c)
    v16bf bR[4], bZ[4], bN[4];
#pragma unroll
    for (int kk = 0; kk < 4; ++kk) {
        bR[kk] = load_frag_b(whhd, (c)      * 16, HDIM, kk * 32);
        bZ[kk] = load_frag_b(whhd, (8 + c)  * 16, HDIM, kk * 32);
        bN[kk] = load_frag_b(whhd, (16 + c) * 16, HDIM, kk * 32);
    }

    // ---- persistent h_prev (fp32) for the 16 elements this lane owns
    float hprev[2][8];
#pragma unroll
    for (int m2 = 0; m2 < 2; ++m2)
#pragma unroll
        for (int v = 0; v < 8; ++v) hprev[m2][v] = 0.f;

    for (int i = tid; i < BSZ * HDIM; i += 512) sHbf[i] = (bf16_t)0.f;
    __syncthreads();

    for (int step = 0; step < TLEN; ++step) {
        const int t = dir ? (TLEN - 1 - step) : step;
        const bf16_t* px = xgd + (size_t)t * (G3 * BSZ);

        // ---- matmul: gh tiles for this wave's slice, A from LDS, B from VGPRs
        v8f aR[2], aZ[2], aN[2];
#pragma unroll
        for (int m2 = 0; m2 < 2; ++m2) {
            aR[m2] = (v8f){0.f,0.f,0.f,0.f,0.f,0.f,0.f,0.f};
            aZ[m2] = (v8f){0.f,0.f,0.f,0.f,0.f,0.f,0.f,0.f};
            aN[m2] = (v8f){0.f,0.f,0.f,0.f,0.f,0.f,0.f,0.f};
            const int mrow = (mh2 * 2 + m2) * 16;
#pragma unroll
            for (int kk = 0; kk < 4; ++kk) {
                v16bf a = load_frag_a(sHbf, mrow, HDIM, kk * 32);
                aR[m2] = WMMA_BF16(a, bR[kk], aR[m2]);
                aZ[m2] = WMMA_BF16(a, bZ[kk], aZ[m2]);
                aN[m2] = WMMA_BF16(a, bN[kk], aN[m2]);
            }
        }
        // prefetch next step's xg gate-chunk rows (each 128B row covers all 64 b)
        if (step + 1 < TLEN) {
            const bf16_t* pn = xgd + (size_t)(dir ? t - 1 : t + 1) * (G3 * BSZ);
            __builtin_prefetch(pn + (size_t)(j)       * BSZ, 0, 1);
            __builtin_prefetch(pn + (size_t)(128 + j) * BSZ, 0, 1);
            __builtin_prefetch(pn + (size_t)(256 + j) * BSZ, 0, 1);
        }
        __syncthreads();   // all WMMA reads of sHbf complete

        // ---- gates in registers: r/z/n for (row m, col j) are in the same lane
#pragma unroll
        for (int m2 = 0; m2 < 2; ++m2) {
            const int mbase = (mh2 * 2 + m2) * 16 + mh;   // first of 8 batch rows
            v8bf xrv = *(const v8bf*)(px + (size_t)(j)       * BSZ + mbase);
            v8bf xzv = *(const v8bf*)(px + (size_t)(128 + j) * BSZ + mbase);
            v8bf xnv = *(const v8bf*)(px + (size_t)(256 + j) * BSZ + mbase);
#pragma unroll
            for (int v = 0; v < 8; ++v) {
                const int b = mbase + v;  // batch row
                float r  = fast_sigmoid((float)xrv[v] + aR[m2][v]);
                float z  = fast_sigmoid((float)xzv[v] + aZ[m2][v]);
                float nn = fast_tanh((float)xnv[v] + r * (aN[m2][v] + bhhn));
                float hv = (1.f - z) * nn + z * hprev[m2][v];
                hprev[m2][v] = hv;
                bf16_t hb = (bf16_t)hv;
                sHbf[b * HDIM + j] = hb;
                hout[((size_t)b * TLEN + t) * 256 + dir * 128 + j] = hb;
            }
        }
        __syncthreads();   // sHbf updated before next step's WMMAs
    }
}

// ---------------- FC head ----------------
__global__ __launch_bounds__(256)
void gru_fc_kernel(const bf16_t* __restrict__ h1,   // [BT][256]
                   const float* __restrict__ fcw,   // [2][256]
                   const float* __restrict__ fcb,   // [2]
                   float* __restrict__ out)         // [BT][2]
{
    int bt = blockIdx.x * blockDim.x + threadIdx.x;
    if (bt >= BT) return;
    const bf16_t* h = h1 + (size_t)bt * 256;
    float a0 = fcb[0], a1 = fcb[1];
#pragma unroll 4
    for (int k = 0; k < 256; ++k) {
        float hv = (float)h[k];
        a0 += hv * fcw[k];
        a1 += hv * fcw[256 + k];
    }
    out[(size_t)bt * 2]     = a0;
    out[(size_t)bt * 2 + 1] = a1;
}

// ---------------- launch ----------------
extern "C" void kernel_launch(void* const* d_in, const int* in_sizes, int n_in,
                              void* d_out, int out_size, void* d_ws, size_t ws_size,
                              hipStream_t stream) {
    const float* x        = (const float*)d_in[0];
    const float* w_ih_l0  = (const float*)d_in[1];
    const float* w_hh_l0  = (const float*)d_in[2];
    const float* b_ih_l0  = (const float*)d_in[3];
    const float* b_hh_l0  = (const float*)d_in[4];
    const float* w_ih_l0r = (const float*)d_in[5];
    const float* w_hh_l0r = (const float*)d_in[6];
    const float* b_ih_l0r = (const float*)d_in[7];
    const float* b_hh_l0r = (const float*)d_in[8];
    const float* w_ih_l1  = (const float*)d_in[9];
    const float* w_hh_l1  = (const float*)d_in[10];
    const float* b_ih_l1  = (const float*)d_in[11];
    const float* b_hh_l1  = (const float*)d_in[12];
    const float* w_ih_l1r = (const float*)d_in[13];
    const float* w_hh_l1r = (const float*)d_in[14];
    const float* b_ih_l1r = (const float*)d_in[15];
    const float* b_hh_l1r = (const float*)d_in[16];
    const float* fc_w     = (const float*)d_in[17];
    const float* fc_b     = (const float*)d_in[18];

    char* ws = (char*)d_ws;
    // workspace layout (bytes)
    bf16_t* xg    = (bf16_t*)(ws);                          // 2*T*384*64*2 = 201326592
    bf16_t* h0    = (bf16_t*)(ws + 201326592ull);           // BT*256*2     =  67108864
    bf16_t* h1    = (bf16_t*)(ws + 268435456ull);           // BT*256*2     =  67108864
    bf16_t* whh0  = (bf16_t*)(ws + 335544320ull);           // 2*384*128*2  =    196608
    bf16_t* whh1  = (bf16_t*)(ws + 335740928ull);           // 2*384*128*2  =    196608
    bf16_t* wih1  = (bf16_t*)(ws + 335937536ull);           // 2*384*256*2  =    393216

    gru_convert_kernel<<<512, 256, 0, stream>>>(w_hh_l0, w_hh_l0r, w_hh_l1, w_hh_l1r,
                                                w_ih_l1, w_ih_l1r, whh0, whh1, wih1);

    gru_proj0_kernel<<<BT, 128, 0, stream>>>(x, w_ih_l0, w_ih_l0r, b_ih_l0, b_ih_l0r,
                                             b_hh_l0, b_hh_l0r, xg);

    gru_rec_kernel<<<2, 512, 0, stream>>>(xg, whh0, b_hh_l0, b_hh_l0r, h0);

    gru_proj1_kernel<<<dim3(BT / 64, 12), 256, 0, stream>>>(h0, wih1, b_ih_l1, b_ih_l1r,
                                                            b_hh_l1, b_hh_l1r, xg);

    gru_rec_kernel<<<2, 512, 0, stream>>>(xg, whh1, b_hh_l1, b_hh_l1r, h1);

    gru_fc_kernel<<<(BT + 255) / 256, 256, 0, stream>>>(h1, fc_w, fc_b, (float*)d_out);
}
